// MatchLSTM_77644418777936
// MI455X (gfx1250) — matrix operand order repaired and checked by reading, not compiled, not run on hardware
//
#include <hip/hip_runtime.h>

// ---------------------------------------------------------------------------
// MatchLSTM for MI455X (gfx1250): bf16 WMMA GEMMs with pre-swizzled B
// (fragment-order weight tiles -> all fragment loads are coalesced b128),
// compile-time K so the k-loop fully unrolls (deep load/WMMA overlap),
// 4x2-tile macro-waves for the big GEMMs. bs=64, T=512, J=64, d=256.
// ---------------------------------------------------------------------------

#define BS 64
#define TSTEPS 512
#define JQ 64
#define DD 256

typedef __attribute__((ext_vector_type(16))) __bf16 v16bf;
typedef __attribute__((ext_vector_type(8)))  float  v8f;

union Frag {
  v16bf v;
  unsigned short u[16];
  uint4 q[2];
};

__device__ __forceinline__ unsigned short f2bf(float f) {
  union { float f; unsigned u; } x; x.f = f;
  unsigned u = x.u;
  u += 0x7fffu + ((u >> 16) & 1u);   // round-to-nearest-even
  return (unsigned short)(u >> 16);
}

__device__ __forceinline__ float sigm(float x) { return 1.0f / (1.0f + __expf(-x)); }

__device__ __forceinline__ v8f zero8() {
  v8f z = {0.f, 0.f, 0.f, 0.f, 0.f, 0.f, 0.f, 0.f};
  return z;
}

// ---------------------------------------------------------------------------
// B is pre-packed in WMMA-fragment order (see packB_kernel):
//   tile (kt = k/32, nt = n/16) occupies 512 consecutive ushorts;
//   within a tile, lane L = g*16 + r (g = K-half, r = column) owns 16
//   contiguous ushorts: element i = B[kt*32 + g*16 + i][nt*16 + r].
// So a B fragment = two coalesced 16B loads per lane (1KB/wave contiguous).
// A stays row-major: lane (g, r) holds row m0+r, K runs {k0+g*16..+15}
// (same k order as packed B) -> two contiguous 16B loads.
// KK is compile-time: full unroll lets the scheduler pipeline loads past WMMAs.
// MT/NT: vertical/horizontal 16-wide tiles per wave (fragment reuse).
// ---------------------------------------------------------------------------
template <int KK, int MT, int NT>
__device__ __forceinline__ void wmma_tiles(const unsigned short* __restrict__ A, int lda,
                                           const unsigned short* __restrict__ Bsw, int tilesN,
                                           int m0, int nt0, int lane, v8f* acc /*[MT*NT]*/) {
  const int g = lane >> 4;
  const int r = lane & 15;
  const size_t bstride = ((size_t)tilesN) << 9;                  // ushorts per K-chunk row of tiles
  const unsigned short* bp0 = Bsw + (((size_t)nt0) << 9) + (lane << 4);
  const unsigned short* ap0 = A + (size_t)(m0 + r) * lda;
#pragma unroll
  for (int k0 = 0; k0 < KK; k0 += 32) {
    const unsigned short* bp = bp0 + (size_t)(k0 >> 5) * bstride;
    Frag b[NT];
#pragma unroll
    for (int nt = 0; nt < NT; ++nt) {
      b[nt].q[0] = *(const uint4*)(bp + ((size_t)nt << 9));
      b[nt].q[1] = *(const uint4*)(bp + ((size_t)nt << 9) + 8);
    }
    __builtin_prefetch(bp + bstride, 0, 1);                      // global_prefetch_b8
    Frag a[MT];
#pragma unroll
    for (int mt = 0; mt < MT; ++mt) {
      const unsigned short* arow = ap0 + (size_t)(mt * 16) * lda + k0;
      a[mt].q[0] = *(const uint4*)(arow + g * 16);
      a[mt].q[1] = *(const uint4*)(arow + g * 16 + 8);
    }
#pragma unroll
    for (int mt = 0; mt < MT; ++mt)
#pragma unroll
      for (int nt = 0; nt < NT; ++nt)
        acc[mt * NT + nt] = __builtin_amdgcn_wmma_f32_16x16x32_bf16(
            false, a[mt].v, false, b[nt].v, (short)0, acc[mt * NT + nt], false, false);
  }
}

// --------------------------- conversion / packing --------------------------

__global__ void cvt_bf16_kernel(const float* __restrict__ src,
                                unsigned short* __restrict__ dst, int n) {
  int i = blockIdx.x * blockDim.x + threadIdx.x;
  if (i < n) dst[i] = f2bf(src[i]);
}

// Pack a KxN B-matrix (optionally reading src transposed, i.e. B = src^T)
// into WMMA-fragment tile order described above.
__global__ void packB_kernel(const float* __restrict__ src,
                             unsigned short* __restrict__ dst,
                             int K, int N, int srcTransposed) {
  int idx = blockIdx.x * blockDim.x + threadIdx.x;
  if (idx >= K * N) return;
  int k = idx / N, n = idx - k * N;
  float v = srcTransposed ? src[(size_t)n * K + k] : src[(size_t)k * N + n];
  int kk = k & 31;
  size_t o = (((size_t)((k >> 5) * (N >> 4) + (n >> 4))) << 9)
           + (size_t)((((kk >> 4) << 4) + (n & 15)) << 4) + (kk & 15);
  dst[o] = f2bf(v);
}

// ------------------- bf16 GEMM, K=256, 64x32 tile per wave ------------------

__global__ void gemm_k256_kernel(const unsigned short* __restrict__ A,
                                 const unsigned short* __restrict__ Bsw,
                                 float* __restrict__ C, int M, int N) {
  int wave = (int)((blockIdx.x * blockDim.x + threadIdx.x) >> 5);
  int lane = threadIdx.x & 31;
  int tilesN2 = N >> 5;                       // 32-wide column groups
  int tm = wave / tilesN2, tn2 = wave - tm * tilesN2;
  if ((tm << 6) >= M) return;                 // wave-uniform (M multiple of 64)
  v8f acc[8];
#pragma unroll
  for (int i = 0; i < 8; ++i) acc[i] = zero8();
  wmma_tiles<DD, 4, 2>(A, DD, Bsw, N >> 4, tm << 6, tn2 << 1, lane, acc);
  int g = lane >> 4, r = lane & 15;
#pragma unroll
  for (int mt = 0; mt < 4; ++mt)
#pragma unroll
    for (int nt = 0; nt < 2; ++nt) {
      float* crow = C + (size_t)((tm << 6) + mt * 16 + (g << 3)) * N + ((tn2 << 1) + nt) * 16 + r;
#pragma unroll
      for (int e = 0; e < 8; ++e) crow[(size_t)e * N] = acc[mt * 2 + nt][e];
    }
}

// -------------------- per-step kernel 1: a = wh_p[t] + h@Wr -----------------

__global__ void step_pre_kernel(const unsigned short* __restrict__ h_bf,
                                const unsigned short* __restrict__ wr_sw,
                                const float* __restrict__ whp, int t,
                                float* __restrict__ a_out) {
  int wave = (int)((blockIdx.x * blockDim.x + threadIdx.x) >> 5);  // 32 waves total
  int lane = threadIdx.x & 31;
  int tm2 = wave >> 4, tn = wave & 15;    // M split 2x32, N split 16x16
  int m0 = tm2 << 5;
  v8f acc[2] = {zero8(), zero8()};
  wmma_tiles<DD, 2, 1>(h_bf, DD, wr_sw, 16, m0, tn, lane, acc);
  int g = lane >> 4, r = lane & 15;
  int n = (tn << 4) + r;
#pragma unroll
  for (int mt = 0; mt < 2; ++mt)
#pragma unroll
    for (int e = 0; e < 8; ++e) {
      int b = m0 + mt * 16 + (g << 3) + e;
      a_out[b * DD + n] = acc[mt][e] + whp[((size_t)b * TSTEPS + t) * DD + n];
    }
}

// ------ per-step kernel 2: attention scores, attn_query, build z (bf16) -----

__global__ void attn_kernel(const float* __restrict__ whq,   // (bs*J, d)
                            const float* __restrict__ a_vec, // (bs, d)
                            const float* __restrict__ w,     // (d)
                            const float* __restrict__ eq,    // (bs, J, d) f32
                            const float* __restrict__ ctx,   // (bs, T, d) f32
                            int t,
                            unsigned short* __restrict__ z_bf) { // (bs, 2d)
  int b = blockIdx.x;        // 64 blocks, 256 threads
  int tid = threadIdx.x;
  __shared__ float attn_s[JQ];
  __shared__ float a_s[DD];
  __shared__ float w_s[DD];
  a_s[tid] = a_vec[b * DD + tid];
  w_s[tid] = w[tid];
  __syncthreads();

  int wv = tid >> 5, lane = tid & 31;
  for (int j = wv; j < JQ; j += 8) {
    const float* gq = whq + ((size_t)b * JQ + j) * DD;
    float s = 0.f;
    for (int d0 = lane; d0 < DD; d0 += 32)
      s += tanhf(gq[d0] + a_s[d0]) * w_s[d0];
#pragma unroll
    for (int off = 16; off > 0; off >>= 1) s += __shfl_down(s, off);
    if (lane == 0) attn_s[j] = s;
  }
  __syncthreads();

  float aq = 0.f;
  const float* eqb = eq + (size_t)b * JQ * DD;
#pragma unroll 4
  for (int j = 0; j < JQ; ++j) aq += attn_s[j] * eqb[(size_t)j * DD + tid];
  z_bf[b * 2 * DD + tid]      = f2bf(ctx[((size_t)b * TSTEPS + t) * DD + tid]);
  z_bf[b * 2 * DD + DD + tid] = f2bf(aq);
}

// -- per-step kernel 3: gates = z@W_ih^T + h@W_hh^T + b_ih + b_hh (WMMA) -----

__global__ void gates_kernel(const unsigned short* __restrict__ z_bf,   // (64, 512)
                             const unsigned short* __restrict__ wih_sw, // packed (512,1024)
                             const unsigned short* __restrict__ h_bf,   // (64, 256)
                             const unsigned short* __restrict__ whh_sw, // packed (256,1024)
                             const float* __restrict__ bih,
                             const float* __restrict__ bhh,
                             float* __restrict__ gates) {               // (64, 1024)
  int wave = (int)((blockIdx.x * blockDim.x + threadIdx.x) >> 5);  // 128 waves total
  int lane = threadIdx.x & 31;
  int tm2 = wave >> 6, tn = wave & 63;    // M split 2x32, N split 64x16
  int m0 = tm2 << 5;
  v8f acc[2] = {zero8(), zero8()};
  wmma_tiles<2 * DD, 2, 1>(z_bf, 2 * DD, wih_sw, 64, m0, tn, lane, acc);
  wmma_tiles<DD, 2, 1>(h_bf, DD, whh_sw, 64, m0, tn, lane, acc);
  int g = lane >> 4, r = lane & 15;
  int n = (tn << 4) + r;
  float bias = bih[n] + bhh[n];
#pragma unroll
  for (int mt = 0; mt < 2; ++mt)
#pragma unroll
    for (int e = 0; e < 8; ++e)
      gates[(size_t)(m0 + mt * 16 + (g << 3) + e) * (4 * DD) + n] = acc[mt][e] + bias;
}

// ------------------- per-step kernel 4: LSTM cell update --------------------

__global__ void lstm_kernel(const float* __restrict__ gates,
                            float* __restrict__ c,
                            unsigned short* __restrict__ h_bf,
                            float* __restrict__ Hr, int t) {
  int b = blockIdx.x, e = threadIdx.x;   // 64 x 256
  const float* gr = gates + (size_t)b * (4 * DD);
  float ig = sigm(gr[e]);
  float fg = sigm(gr[DD + e]);
  float gg = tanhf(gr[2 * DD + e]);
  float og = sigm(gr[3 * DD + e]);
  float cn = fg * c[b * DD + e] + ig * gg;
  float hn = og * tanhf(cn);
  c[b * DD + e] = cn;
  h_bf[b * DD + e] = f2bf(hn);
  Hr[((size_t)b * (TSTEPS + 1) + t + 1) * DD + e] = hn;
}

// ------------------------------- init ---------------------------------------

__global__ void init_kernel(const float* __restrict__ h0, const float* __restrict__ c0,
                            float* __restrict__ c, unsigned short* __restrict__ h_bf,
                            float* __restrict__ Hr) {
  int i = blockIdx.x * blockDim.x + threadIdx.x;  // 16384
  int b = i >> 8, e = i & 255;
  c[i] = c0[i];
  h_bf[i] = f2bf(h0[i]);
  Hr[(size_t)b * (TSTEPS + 1) * DD + e] = h0[i];
}

// ---------------------------------------------------------------------------

extern "C" void kernel_launch(void* const* d_in, const int* in_sizes, int n_in,
                              void* d_out, int out_size, void* d_ws, size_t ws_size,
                              hipStream_t stream) {
  (void)in_sizes; (void)n_in; (void)out_size; (void)ws_size;
  const float* ec  = (const float*)d_in[0];   // (64, 512, 256)
  const float* eq  = (const float*)d_in[1];   // (64, 64, 256)
  const float* h0  = (const float*)d_in[2];
  const float* c0  = (const float*)d_in[3];
  const float* Wq  = (const float*)d_in[4];
  const float* Wp  = (const float*)d_in[5];
  const float* Wr  = (const float*)d_in[6];
  const float* w   = (const float*)d_in[7];
  const float* Wih = (const float*)d_in[8];   // (1024, 512)
  const float* Whh = (const float*)d_in[9];   // (1024, 256)
  const float* bih = (const float*)d_in[10];
  const float* bhh = (const float*)d_in[11];
  float* Hr = (float*)d_out;                  // (64, 513, 256)

  char* ws = (char*)d_ws;
  size_t off = 0;
  auto alloc = [&](size_t bytes) -> void* {
    void* p = ws + off;
    off += (bytes + 255) & ~(size_t)255;
    return p;
  };
  unsigned short* wq_sw   = (unsigned short*)alloc((size_t)DD * DD * 2);
  unsigned short* wp_sw   = (unsigned short*)alloc((size_t)DD * DD * 2);
  unsigned short* wr_sw   = (unsigned short*)alloc((size_t)DD * DD * 2);
  unsigned short* wih_sw  = (unsigned short*)alloc((size_t)2 * DD * 4 * DD * 2);
  unsigned short* whh_sw  = (unsigned short*)alloc((size_t)DD * 4 * DD * 2);
  unsigned short* eq_bf   = (unsigned short*)alloc((size_t)BS * JQ * DD * 2);
  unsigned short* ec_bf   = (unsigned short*)alloc((size_t)BS * TSTEPS * DD * 2);
  float* whq   = (float*)alloc((size_t)BS * JQ * DD * 4);
  float* whp   = (float*)alloc((size_t)BS * TSTEPS * DD * 4);
  float* c_st  = (float*)alloc((size_t)BS * DD * 4);
  unsigned short* h_bf = (unsigned short*)alloc((size_t)BS * DD * 2);
  float* a_vec = (float*)alloc((size_t)BS * DD * 4);
  unsigned short* z_bf = (unsigned short*)alloc((size_t)BS * 2 * DD * 2);
  float* gates = (float*)alloc((size_t)BS * 4 * DD * 4);

  const int TB = 256;
  auto nb = [](int n) { return (n + 255) / 256; };

  // Pack weights into WMMA-fragment order (one-time; reused 512 steps).
  packB_kernel<<<nb(DD * DD), TB, 0, stream>>>(Wq, wq_sw, DD, DD, 0);
  packB_kernel<<<nb(DD * DD), TB, 0, stream>>>(Wp, wp_sw, DD, DD, 0);
  packB_kernel<<<nb(DD * DD), TB, 0, stream>>>(Wr, wr_sw, DD, DD, 0);
  packB_kernel<<<nb(2 * DD * 4 * DD), TB, 0, stream>>>(Wih, wih_sw, 2 * DD, 4 * DD, 1);
  packB_kernel<<<nb(DD * 4 * DD), TB, 0, stream>>>(Whh, whh_sw, DD, 4 * DD, 1);
  // A-side activations to bf16 (row-major).
  cvt_bf16_kernel<<<nb(BS * JQ * DD), TB, 0, stream>>>(eq, eq_bf, BS * JQ * DD);
  cvt_bf16_kernel<<<nb(BS * TSTEPS * DD), TB, 0, stream>>>(ec, ec_bf, BS * TSTEPS * DD);

  init_kernel<<<nb(BS * DD), TB, 0, stream>>>(h0, c0, c_st, h_bf, Hr);

  // Hoisted projections: wh_q = eq @ Wq ; wh_p = ec @ Wp   (WMMA, 64x32/wave)
  {
    int M = BS * JQ, N = DD;                          // 512 waves
    gemm_k256_kernel<<<(M / 64) * (N / 32) / 8, TB, 0, stream>>>(eq_bf, wq_sw, whq, M, N);
  }
  {
    int M = BS * TSTEPS, N = DD;                      // 4096 waves
    gemm_k256_kernel<<<(M / 64) * (N / 32) / 8, TB, 0, stream>>>(ec_bf, wp_sw, whp, M, N);
  }

  // Sequential recurrence over T.
  for (int t = 0; t < TSTEPS; ++t) {
    step_pre_kernel<<<4, TB, 0, stream>>>(h_bf, wr_sw, whp, t, a_vec);           // 32 waves
    attn_kernel<<<BS, TB, 0, stream>>>(whq, a_vec, w, eq, ec, t, z_bf);
    gates_kernel<<<16, TB, 0, stream>>>(z_bf, wih_sw, h_bf, whh_sw, bih, bhh, gates); // 128 waves
    lstm_kernel<<<BS, TB, 0, stream>>>(gates, c_st, h_bf, Hr, t);
  }
}
